// VVITLayer_78142634983466
// MI455X (gfx1250) — compile-verified, matrix operand order checked by reading, not run on hardware
//
#include <hip/hip_runtime.h>
#include <hip/hip_bf16.h>

// ---------------------------------------------------------------------------
// VVIT layer for MI455X (gfx1250, wave32, WMMA).
// Dims fixed by reference: B=4, L=8, N=2048, c=256, H=8, D=32, rows=B*N=8192.
// All GEMMs use v_wmma_f32_16x16x32_f16 (f16 in, f32 accumulate).
// Attention is flash-style (online softmax): the 537MB att matrix is never
// materialized. K/V chunks are staged into LDS once per 4-wave workgroup with
// double-buffered GLOBAL_LOAD_ASYNC_TO_LDS_B128 (ASYNCcnt) so the DMA overlaps
// the WMMA+softmax work and L2 traffic drops 4x vs per-wave streaming.
// ---------------------------------------------------------------------------

typedef __attribute__((ext_vector_type(16))) _Float16 v16h;
typedef __attribute__((ext_vector_type(8)))  float    v8f;

#define WMMA_F16(a, b, c) \
  __builtin_amdgcn_wmma_f32_16x16x32_f16(false, (a), false, (b), (short)0, (c), false, false)

// Load a 16x32 f16 A-fragment (or B-fragment fed as B^T) from a row-major
// [16 x stride] region. CDNA5 ISA layout: lanes 0-15 hold row=lane with
// K in {0..7, 16..23}; lanes 16-31 hold row=lane-16 with K in {8..15, 24..31}.
static __device__ __forceinline__ v16h load_frag_a(const _Float16* __restrict__ base,
                                                   int stride) {
  const int lane = threadIdx.x & 31;
  const int row  = lane & 15;
  const int koff = (lane >> 4) << 3;   // 0 or 8
  const _Float16* p = base + row * stride + koff;
  v16h a;
  ((float4*)&a)[0] = *(const float4*)(p);        // k = koff .. koff+7   -> VGPR0-3
  ((float4*)&a)[1] = *(const float4*)(p + 16);   // k = koff+16..koff+23 -> VGPR4-7
  return a;
}

// Async DMA: 16 bytes global -> LDS, tracked by ASYNCcnt (no VGPR data path).
// VDST holds the wave-relative LDS byte address (= low 32 bits of the generic
// shared pointer, per the CDNA5 LDS aperture rule addr[31:0]).
static __device__ __forceinline__ void async_copy_b128(void* lds_dst,
                                                       const _Float16* gsrc) {
  const unsigned lds_off = (unsigned)(uintptr_t)lds_dst;
  const unsigned long long ga = (unsigned long long)(uintptr_t)gsrc;
  asm volatile("global_load_async_to_lds_b128 %0, %1, off"
               :: "v"(lds_off), "v"(ga)
               : "memory");
}

static __device__ __forceinline__ void wait_async0() {
  asm volatile("s_wait_asynccnt 0x0" ::: "memory");
}

// ---------------------------------------------------------------------------
// K0: weight prep: f32 -> f16, transposed so WMMA B-fragments load row-major.
//   WhT [H=8][D=32][c=256]  = Wh[h][c][d]
//   fc1T[out=256][c=256]    = fc1_w[c][out]
//   fc2T[cout=256][hid=256] = fc2_w[hid][cout]
// ---------------------------------------------------------------------------
__global__ void k_prep(const float* __restrict__ Wh,
                       const float* __restrict__ fc1w,
                       const float* __restrict__ fc2w,
                       _Float16* __restrict__ WhT,
                       _Float16* __restrict__ fc1T,
                       _Float16* __restrict__ fc2T) {
  const int i = blockIdx.x * blockDim.x + threadIdx.x;   // 0..65535
  if (i >= 65536) return;
  const int h = i >> 13, rem = i & 8191, d = rem >> 8, ch = rem & 255;
  WhT[i] = (_Float16)Wh[(h * 256 + ch) * 32 + d];
  const int o = i >> 8, c2 = i & 255;
  fc1T[i] = (_Float16)fc1w[c2 * 256 + o];
  fc2T[i] = (_Float16)fc2w[c2 * 256 + o];
}

// ---------------------------------------------------------------------------
// LayerNorm over c=256. One block (256 threads) per row. gather==1 reads from
// x[b,c,h,w] with the [B,L,hw,c] transpose; gather==0 reads row-major [row,c].
// Writes both f32 (for residual) and f16 (for WMMA A-fragments).
// ---------------------------------------------------------------------------
__global__ __launch_bounds__(256) void k_ln(const float* __restrict__ in,
                                            const float* __restrict__ g,
                                            const float* __restrict__ bb,
                                            float* __restrict__ o32,
                                            _Float16* __restrict__ o16,
                                            int gather) {
  const int row = blockIdx.x;          // b*2048 + n
  const int ch  = threadIdx.x;
  float v;
  if (gather) {
    const int b = row >> 11, n = row & 2047, l = n >> 8, p = n & 255;
    v = in[(((size_t)(b * 8 + l) * 256 + ch) * 256) + p];
  } else {
    v = in[(size_t)row * 256 + ch];
  }
  __shared__ float s1[256], s2[256];
  s1[ch] = v; s2[ch] = v * v;
  __syncthreads();
  for (int off = 128; off > 0; off >>= 1) {
    if (ch < off) { s1[ch] += s1[ch + off]; s2[ch] += s2[ch + off]; }
    __syncthreads();
  }
  const float mean = s1[0] * (1.0f / 256.0f);
  const float var  = s2[0] * (1.0f / 256.0f) - mean * mean;   // biased, as torch LN
  const float inv  = rsqrtf(var + 1e-5f);
  const float o = (v - mean) * inv * g[ch] + bb[ch];
  o32[(size_t)row * 256 + ch] = o;
  o16[(size_t)row * 256 + ch] = (_Float16)o;
}

// ---------------------------------------------------------------------------
// K2: head projection xh = t @ Wh + bh. One wave per 16x16 output tile,
// K-loop over c=256 (8 WMMAs). Writes xh [B,H,N,32] (Q/K source) and
// xhT [B,H,32,N] (V^T source for the P*V WMMA).
// ---------------------------------------------------------------------------
__global__ __launch_bounds__(128) void k_qkv(const _Float16* __restrict__ t16,
                                             const _Float16* __restrict__ WhT,
                                             const float* __restrict__ bh,
                                             _Float16* __restrict__ xh,
                                             _Float16* __restrict__ xhT) {
  const int wid  = (blockIdx.x * blockDim.x + threadIdx.x) >> 5;  // 0..8191
  const int lane = threadIdx.x & 31;
  const int dt = wid & 1;            // which 16-wide half of D=32
  const int qt = (wid >> 1) & 127;   // row tile
  const int h  = (wid >> 8) & 7;
  const int b  = wid >> 11;

  const _Float16* abase = t16 + (size_t)(b * 2048 + qt * 16) * 256;
  const _Float16* bbase = WhT + (size_t)(h * 32 + dt * 16) * 256;
  v8f acc = {0.f, 0.f, 0.f, 0.f, 0.f, 0.f, 0.f, 0.f};
#pragma unroll
  for (int k = 0; k < 256; k += 32) {
    v16h a  = load_frag_a(abase + k, 256);
    v16h bf = load_frag_a(bbase + k, 256);
    acc = WMMA_F16(a, bf, acc);
  }
  const int col   = lane & 15;
  const int mbase = (lane >> 4) << 3;         // C-layout: lanes>=16 hold M=8..15
  const int d     = dt * 16 + col;
  const float bias = bh[h * 32 + d];
  _Float16* xo  = xh  + (size_t)(b * 8 + h) * 2048 * 32;
  _Float16* xto = xhT + (size_t)(b * 8 + h) * 32 * 2048;
#pragma unroll
  for (int r = 0; r < 8; ++r) {
    const int n = qt * 16 + mbase + r;
    const _Float16 hv = (_Float16)(acc[r] + bias);
    xo[(size_t)n * 32 + d]    = hv;
    xto[(size_t)d * 2048 + n] = hv;
  }
}

// ---------------------------------------------------------------------------
// K3: flash attention + relu + residual.
// Block = 4 waves, all on the same (b,h); wave wv owns q-tile qt0+wv (16 rows).
// kv loop in chunks of 32: the 32x32 K chunk and 32x32 V^T chunk are staged
// into double-buffered LDS by GLOBAL_LOAD_ASYNC_TO_LDS_B128 (one 16B segment
// per thread), overlapped with the previous chunk's 2 QK WMMAs + online
// softmax + 2 PV WMMAs. P tiles convert C-layout -> A-layout via padded LDS.
// Writes y[b,n,c] = t + relu(O/l) at channels [h*32, h*32+32).
// ---------------------------------------------------------------------------
__global__ __launch_bounds__(128) void k_attn(const _Float16* __restrict__ xh,
                                              const _Float16* __restrict__ xhT,
                                              const float* __restrict__ t32,
                                              float* __restrict__ y) {
  const int lane = threadIdx.x & 31;
  const int wv   = threadIdx.x >> 5;           // wave in block: 0..3
  const int blk  = blockIdx.x;                 // 0..1023
  const int qt   = (blk & 31) * 4 + wv;        // q tile 0..127
  const int h    = (blk >> 5) & 7;
  const int b    = blk >> 8;

  __shared__ _Float16 Kb[2][32][40];           // K rows chunk   (kv x d)
  __shared__ _Float16 Vb[2][32][40];           // V^T rows chunk (d x kv)
  __shared__ _Float16 Pl[4][16][40];           // per-wave P tile

  const size_t head = (size_t)(b * 8 + h);
  const _Float16* kbase  = xh  + head * 2048 * 32;
  const _Float16* vtbase = xhT + head * 32 * 2048;

  // async staging: thread t copies 16B segment seg of row (t>>2) of each chunk
  const int srow = threadIdx.x >> 2;           // 0..31
  const int sseg = (threadIdx.x & 3) << 3;     // 0,8,16,24 halves

  // preload chunk 0
  async_copy_b128(&Kb[0][srow][sseg], kbase + (size_t)srow * 32 + sseg);
  async_copy_b128(&Vb[0][srow][sseg], vtbase + (size_t)srow * 2048 + sseg);

  const v16h aq = load_frag_a(xh + (head * 2048 + qt * 16) * 32, 32);

  v8f o0 = {0.f, 0.f, 0.f, 0.f, 0.f, 0.f, 0.f, 0.f};
  v8f o1 = o0;
  const v8f zero = o0;
  float m[8], lsum[8];
#pragma unroll
  for (int r = 0; r < 8; ++r) { m[r] = -3.0e38f; lsum[r] = 0.f; }

  const int col   = lane & 15;
  const int mbase = (lane >> 4) << 3;

  for (int it = 0; it < 64; ++it) {
    const int buf = it & 1;
    wait_async0();                 // this wave's staged segments are in LDS
    __syncthreads();               // whole block's segments are in LDS
    if (it + 1 < 64) {             // kick DMA for the next chunk (other buffer)
      const int kv1 = (it + 1) * 32;
      async_copy_b128(&Kb[buf ^ 1][srow][sseg],
                      kbase + (size_t)(kv1 + srow) * 32 + sseg);
      async_copy_b128(&Vb[buf ^ 1][srow][sseg],
                      vtbase + (size_t)srow * 2048 + kv1 + sseg);
    }

    // S tile: Q(16x32) x K^T ; B-fragment == rows of K from LDS
    const v16h bk0 = load_frag_a(&Kb[buf][0][0], 40);
    const v16h bk1 = load_frag_a(&Kb[buf][16][0], 40);
    v8f s0 = WMMA_F16(aq, bk0, zero);
    v8f s1 = WMMA_F16(aq, bk1, zero);

    // row max over the 32 kv columns (reduce across 16-lane halves)
    float rmax[8];
#pragma unroll
    for (int r = 0; r < 8; ++r) rmax[r] = fmaxf(s0[r], s1[r]);
#pragma unroll
    for (int msk = 1; msk < 16; msk <<= 1)
#pragma unroll
      for (int r = 0; r < 8; ++r) rmax[r] = fmaxf(rmax[r], __shfl_xor(rmax[r], msk, 32));

    float p0[8], p1[8], rs[8];
#pragma unroll
    for (int r = 0; r < 8; ++r) {
      const float mn = fmaxf(m[r], rmax[r]);
      const float sc = __expf(m[r] - mn);
      m[r] = mn;
      o0[r] *= sc; o1[r] *= sc;
      p0[r] = __expf(s0[r] - mn);
      p1[r] = __expf(s1[r] - mn);
      rs[r] = p0[r] + p1[r];
      lsum[r] *= sc;
    }
#pragma unroll
    for (int msk = 1; msk < 16; msk <<= 1)
#pragma unroll
      for (int r = 0; r < 8; ++r) rs[r] += __shfl_xor(rs[r], msk, 32);
#pragma unroll
    for (int r = 0; r < 8; ++r) lsum[r] += rs[r];

    // C-layout -> LDS [16 rows][32 kv cols]
#pragma unroll
    for (int r = 0; r < 8; ++r) {
      Pl[wv][mbase + r][col]      = (_Float16)p0[r];
      Pl[wv][mbase + r][16 + col] = (_Float16)p1[r];
    }
    __syncthreads();
    const v16h ap  = load_frag_a(&Pl[wv][0][0], 40);
    const v16h bv0 = load_frag_a(&Vb[buf][0][0], 40);   // V^T rows d=0..15
    const v16h bv1 = load_frag_a(&Vb[buf][16][0], 40);  // V^T rows d=16..31
    o0 = WMMA_F16(ap, bv0, o0);
    o1 = WMMA_F16(ap, bv1, o1);
    __syncthreads();   // compute done before next iter's DMA overwrites buf^1
  }

  // finalize: O/l, relu, + residual t; channels h*32 + [0,32)
#pragma unroll
  for (int r = 0; r < 8; ++r) {
    const float inv = 1.0f / lsum[r];
    const int n = qt * 16 + mbase + r;
    const size_t base = ((size_t)b * 2048 + n) * 256 + h * 32;
    y[base + col]      = t32[base + col]      + fmaxf(o0[r] * inv, 0.f);
    y[base + 16 + col] = t32[base + 16 + col] + fmaxf(o1[r] * inv, 0.f);
  }
}

// ---------------------------------------------------------------------------
// K5: hmid = relu(z @ fc1_w + fc1_b), [8192x256]@[256x256], WMMA tiles.
// ---------------------------------------------------------------------------
__global__ __launch_bounds__(128) void k_fc1(const _Float16* __restrict__ z16,
                                             const _Float16* __restrict__ fc1T,
                                             const float* __restrict__ fc1b,
                                             _Float16* __restrict__ hmid) {
  const int wid  = (blockIdx.x * blockDim.x + threadIdx.x) >> 5;  // 0..8191
  const int lane = threadIdx.x & 31;
  const int ct = wid & 15;      // output col tile
  const int rt = wid >> 4;      // row tile (0..511)
  const _Float16* abase = z16  + (size_t)rt * 16 * 256;
  const _Float16* bbase = fc1T + (size_t)ct * 16 * 256;
  v8f acc = {0.f, 0.f, 0.f, 0.f, 0.f, 0.f, 0.f, 0.f};
#pragma unroll
  for (int k = 0; k < 256; k += 32)
    acc = WMMA_F16(load_frag_a(abase + k, 256), load_frag_a(bbase + k, 256), acc);
  const int col = lane & 15, mbase = (lane >> 4) << 3;
  const float bias = fc1b[ct * 16 + col];
#pragma unroll
  for (int r = 0; r < 8; ++r) {
    const int row = rt * 16 + mbase + r;
    hmid[(size_t)row * 256 + ct * 16 + col] = (_Float16)fmaxf(acc[r] + bias, 0.f);
  }
}

// ---------------------------------------------------------------------------
// K6: out = z + relu(hmid @ fc2_w + fc2_b), scattered back to [b,c,h,w].
// ---------------------------------------------------------------------------
__global__ __launch_bounds__(128) void k_fc2(const _Float16* __restrict__ hmid,
                                             const _Float16* __restrict__ fc2T,
                                             const float* __restrict__ fc2b,
                                             const float* __restrict__ z32,
                                             float* __restrict__ out) {
  const int wid  = (blockIdx.x * blockDim.x + threadIdx.x) >> 5;
  const int lane = threadIdx.x & 31;
  const int ct = wid & 15;
  const int rt = wid >> 4;
  const _Float16* abase = hmid + (size_t)rt * 16 * 256;
  const _Float16* bbase = fc2T + (size_t)ct * 16 * 256;
  v8f acc = {0.f, 0.f, 0.f, 0.f, 0.f, 0.f, 0.f, 0.f};
#pragma unroll
  for (int k = 0; k < 256; k += 32)
    acc = WMMA_F16(load_frag_a(abase + k, 256), load_frag_a(bbase + k, 256), acc);
  const int col = lane & 15, mbase = (lane >> 4) << 3;
  const int ch = ct * 16 + col;
  const float bias = fc2b[ch];
#pragma unroll
  for (int r = 0; r < 8; ++r) {
    const int row = rt * 16 + mbase + r;          // b*2048 + n
    const float val = z32[(size_t)row * 256 + ch] + fmaxf(acc[r] + bias, 0.f);
    const int b = row >> 11, n = row & 2047, l = n >> 8, p = n & 255;
    out[(((size_t)(b * 8 + l) * 256 + ch) * 256) + p] = val;
  }
}

// ---------------------------------------------------------------------------
extern "C" void kernel_launch(void* const* d_in, const int* in_sizes, int n_in,
                              void* d_out, int out_size, void* d_ws, size_t ws_size,
                              hipStream_t stream) {
  const float* x      = (const float*)d_in[0];
  const float* Wh     = (const float*)d_in[1];
  const float* bh     = (const float*)d_in[2];
  const float* norm_g = (const float*)d_in[3];
  const float* norm_b = (const float*)d_in[4];
  const float* fn_g   = (const float*)d_in[5];
  const float* fn_b   = (const float*)d_in[6];
  const float* fc1_w  = (const float*)d_in[7];
  const float* fc1_b  = (const float*)d_in[8];
  const float* fc2_w  = (const float*)d_in[9];
  const float* fc2_b  = (const float*)d_in[10];
  float* out = (float*)d_out;

  char* ws = (char*)d_ws;
  size_t off = 0;
  auto alloc = [&](size_t bytes) -> void* {
    void* p = ws + off;
    off += (bytes + 255) & ~(size_t)255;
    return p;
  };
  const size_t ROWS = 8192;                   // B*N
  float*    tf32 = (float*)   alloc(ROWS * 256 * 4);
  float*    yb   = (float*)   alloc(ROWS * 256 * 4);
  float*    z32  = (float*)   alloc(ROWS * 256 * 4);
  _Float16* tf16 = (_Float16*)alloc(ROWS * 256 * 2);
  _Float16* zf16 = (_Float16*)alloc(ROWS * 256 * 2);
  _Float16* xh   = (_Float16*)alloc((size_t)4 * 8 * 2048 * 32 * 2);
  _Float16* xhT  = (_Float16*)alloc((size_t)4 * 8 * 2048 * 32 * 2);
  _Float16* hmid = (_Float16*)alloc(ROWS * 256 * 2);
  _Float16* WhT  = (_Float16*)alloc(65536 * 2);
  _Float16* fc1T = (_Float16*)alloc(65536 * 2);
  _Float16* fc2T = (_Float16*)alloc(65536 * 2);

  k_prep<<<256, 256, 0, stream>>>(Wh, fc1_w, fc2_w, WhT, fc1T, fc2T);
  k_ln  <<<8192, 256, 0, stream>>>(x, norm_g, norm_b, tf32, tf16, 1);
  k_qkv <<<2048, 128, 0, stream>>>(tf16, WhT, bh, xh, xhT);
  k_attn<<<1024, 128, 0, stream>>>(xh, xhT, tf32, yb);
  k_ln  <<<8192, 256, 0, stream>>>(yb, fn_g, fn_b, z32, zf16, 0);
  k_fc1 <<<2048, 128, 0, stream>>>(zf16, fc1T, fc1_b, hmid);
  k_fc2 <<<2048, 128, 0, stream>>>(hmid, fc2T, fc2_b, z32, out);
}
